// LargeMarginLoss_22454089023818
// MI455X (gfx1250) — matrix operand order repaired and checked by reading, not compiled, not run on hardware
//
#include <hip/hip_runtime.h>
#include <hip/hip_fp16.h>
#include <math.h>

typedef __attribute__((ext_vector_type(16))) _Float16 v16h;
typedef __attribute__((ext_vector_type(8)))  _Float16 v8h;
typedef __attribute__((ext_vector_type(8)))  float    v8f;

#define BN 8192      // batch
#define DK 512       // embedding dim
#define NSTEP (DK / 32)   // 16 k-slabs
#define SCALE_F 30.0f
#define MARGIN_F 0.2f
#define LDS_STRIDE 40   // 32 halfs data + 8 pad -> 80B row stride, bank-conflict free

// ---------------- Kernel 1: row normalize, f32 -> f16 ----------------
__global__ void __launch_bounds__(256)
norm_rows(const float* __restrict__ E, _Float16* __restrict__ Nh) {
    __shared__ float red[256];
    const int row = blockIdx.x;
    const float* er = E + (size_t)row * DK;
    float v0 = er[threadIdx.x];
    float v1 = er[threadIdx.x + 256];
    red[threadIdx.x] = v0 * v0 + v1 * v1;
    __syncthreads();
    for (int s = 128; s > 0; s >>= 1) {
        if (threadIdx.x < s) red[threadIdx.x] += red[threadIdx.x + s];
        __syncthreads();
    }
    const float scale = 1.0f / fmaxf(sqrtf(red[0]), 1e-8f);
    _Float16* nr = Nh + (size_t)row * DK;
    nr[threadIdx.x]       = (_Float16)(v0 * scale);
    nr[threadIdx.x + 256] = (_Float16)(v1 * scale);
}

// ------------- Kernel 2: fused N*N^T + exp + masked reduce -----------
// Grid 64x64, 256 threads (8 waves). Block tile 128x128.
// Wave w: rows (w>>1)*32, cols (w&1)*64 -> 2x4 WMMA tiles of 16x16.
// Double-buffered LDS, register-prefetched global loads, one barrier/slab.
__global__ void __launch_bounds__(256)
sim_exp_gemm(const _Float16* __restrict__ Nh, const int* __restrict__ lab,
             float* __restrict__ partial) {
    __shared__ _Float16 As[2][128 * LDS_STRIDE];
    __shared__ _Float16 Bs[2][128 * LDS_STRIDE];
    __shared__ int   rowLab[128];
    __shared__ int   colLab[128];
    __shared__ float redp[256];
    __shared__ float redn[256];

    const int t       = threadIdx.x;
    const int lane    = t & 31;
    const int wave    = t >> 5;
    const int rowBase = blockIdx.y * 128;
    const int colBase = blockIdx.x * 128;
    const int waveRow = (wave >> 1) * 32;
    const int waveCol = (wave & 1) * 64;
    const int lane15  = lane & 15;
    const int halfSel = lane >> 4;     // 0: lanes 0-15, 1: lanes 16-31

    if (t < 128) rowLab[t] = lab[rowBase + t];
    else         colLab[t - 128] = lab[colBase + (t - 128)];

    // ---- hoisted staging addresses: thread t owns rows r0 and r0+64,
    //      16B column p, for both the A (row) tile and B (col) tile ----
    const int r0 = t >> 2;            // 0..63
    const int p  = (t & 3) * 8;       // half offset: 0,8,16,24
    const _Float16* gA0 = Nh + (size_t)(rowBase + r0)      * DK + p;
    const _Float16* gA1 = Nh + (size_t)(rowBase + r0 + 64) * DK + p;
    const _Float16* gB0 = Nh + (size_t)(colBase + r0)      * DK + p;
    const _Float16* gB1 = Nh + (size_t)(colBase + r0 + 64) * DK + p;
    const int sOff0 = r0 * LDS_STRIDE + p;
    const int sOff1 = (r0 + 64) * LDS_STRIDE + p;

    v8f acc[2][4];
    v8f zero = {};
#pragma unroll
    for (int mi = 0; mi < 2; ++mi)
#pragma unroll
        for (int ni = 0; ni < 4; ++ni) acc[mi][ni] = zero;

    // preload slab 0 into buffer 0
    {
        v8h ra0 = *(const v8h*)gA0;
        v8h ra1 = *(const v8h*)gA1;
        v8h rb0 = *(const v8h*)gB0;
        v8h rb1 = *(const v8h*)gB1;
        *(v8h*)&As[0][sOff0] = ra0;
        *(v8h*)&As[0][sOff1] = ra1;
        *(v8h*)&Bs[0][sOff0] = rb0;
        *(v8h*)&Bs[0][sOff1] = rb1;
    }

#pragma unroll
    for (int ks = 0; ks < NSTEP; ++ks) {
        const int bi = ks & 1;
        __syncthreads();   // slab ks visible; prior reads of buffer bi^1 done

        // A fragments: lane<16 holds row M=lane15, K slots {0..7,16..23};
        // lane>=16 holds K slots {8..15,24..31}.
        v16h a[2];
#pragma unroll
        for (int mi = 0; mi < 2; ++mi) {
            const int r  = waveRow + mi * 16 + lane15;
            const int kb = halfSel * 8;
            v8h lo = *(const v8h*)&As[bi][r * LDS_STRIDE + kb];
            v8h hi = *(const v8h*)&As[bi][r * LDS_STRIDE + kb + 16];
            a[mi] = __builtin_shufflevector(lo, hi, 0,1,2,3,4,5,6,7,
                                            8,9,10,11,12,13,14,15);
        }
        // B fragments: column j of B is row j of N. lane<16: K {0..7,16..23},
        // lane>=16: K {8..15,24..31}, matching A.
        v16h b[4];
#pragma unroll
        for (int ni = 0; ni < 4; ++ni) {
            const int cix = waveCol + ni * 16 + lane15;
            const int kb  = halfSel * 8;
            v8h lo = *(const v8h*)&Bs[bi][cix * LDS_STRIDE + kb];
            v8h hi = *(const v8h*)&Bs[bi][cix * LDS_STRIDE + kb + 16];
            b[ni] = __builtin_shufflevector(lo, hi, 0,1,2,3,4,5,6,7,
                                            8,9,10,11,12,13,14,15);
        }

        // register-prefetch next slab from global (overlaps with WMMA)
        v8h ra0, ra1, rb0, rb1;
        if (ks + 1 < NSTEP) {
            const int ko = (ks + 1) * 32;
            ra0 = *(const v8h*)(gA0 + ko);
            ra1 = *(const v8h*)(gA1 + ko);
            rb0 = *(const v8h*)(gB0 + ko);
            rb1 = *(const v8h*)(gB1 + ko);
        }

#pragma unroll
        for (int mi = 0; mi < 2; ++mi)
#pragma unroll
            for (int ni = 0; ni < 4; ++ni)
                acc[mi][ni] = __builtin_amdgcn_wmma_f32_16x16x32_f16(
                    false, a[mi], false, b[ni], (short)0, acc[mi][ni],
                    false, false);

        // stage next slab into the other buffer
        if (ks + 1 < NSTEP) {
            const int bn = bi ^ 1;
            *(v8h*)&As[bn][sOff0] = ra0;
            *(v8h*)&As[bn][sOff1] = ra1;
            *(v8h*)&Bs[bn][sOff0] = rb0;
            *(v8h*)&Bs[bn][sOff1] = rb1;
        }
    }

    // Fused epilogue: C layout -> lanes 0-15 hold N=lane, M=vgpr;
    // lanes 16-31 hold N=lane-16, M=vgpr+8.
    float pos = 0.0f, neg = 0.0f;
#pragma unroll
    for (int mi = 0; mi < 2; ++mi) {
#pragma unroll
        for (int ni = 0; ni < 4; ++ni) {
            const int cl = colLab[waveCol + ni * 16 + lane15];
#pragma unroll
            for (int v = 0; v < 8; ++v) {
                const int rowL = waveRow + mi * 16 + v + halfSel * 8;
                const float e = __expf(SCALE_F * acc[mi][ni][v]
                                       - SCALE_F * MARGIN_F);
                if (rowLab[rowL] == cl) pos += e; else neg += e;
            }
        }
    }
    redp[t] = pos; redn[t] = neg;
    __syncthreads();
    for (int s = 128; s > 0; s >>= 1) {
        if (t < s) { redp[t] += redp[t + s]; redn[t] += redn[t + s]; }
        __syncthreads();
    }
    if (t == 0) {
        const int bid = blockIdx.y * gridDim.x + blockIdx.x;
        partial[2 * bid]     = redp[0];
        partial[2 * bid + 1] = redn[0];
    }
}

// ---------------- Kernel 3: final reduce + log1p ---------------------
__global__ void __launch_bounds__(256)
finalize(const float* __restrict__ partial, int nblocks, float* __restrict__ out) {
    __shared__ float rp[256], rn[256];
    float p = 0.0f, n = 0.0f;
    for (int i = threadIdx.x; i < nblocks; i += 256) {
        p += partial[2 * i];
        n += partial[2 * i + 1];
    }
    rp[threadIdx.x] = p; rn[threadIdx.x] = n;
    __syncthreads();
    for (int s = 128; s > 0; s >>= 1) {
        if (threadIdx.x < s) {
            rp[threadIdx.x] += rp[threadIdx.x + s];
            rn[threadIdx.x] += rn[threadIdx.x + s];
        }
        __syncthreads();
    }
    if (threadIdx.x == 0) out[0] = log1pf(rn[0] / rp[0]);
}

extern "C" void kernel_launch(void* const* d_in, const int* in_sizes, int n_in,
                              void* d_out, int out_size, void* d_ws, size_t ws_size,
                              hipStream_t stream) {
    const float* E   = (const float*)d_in[0];   // [8192, 512] f32
    const int*   lab = (const int*)d_in[1];     // [8192] int
    float*       out = (float*)d_out;           // scalar f32

    _Float16* Nh      = (_Float16*)d_ws;                             // 8 MB
    float*    partial = (float*)((char*)d_ws + (size_t)BN * DK * 2); // 32 KB

    norm_rows<<<BN, 256, 0, stream>>>(E, Nh);

    dim3 grid(BN / 128, BN / 128);   // 64 x 64
    sim_exp_gemm<<<grid, 256, 0, stream>>>(Nh, lab, partial);

    finalize<<<1, 256, 0, stream>>>(partial, (BN / 128) * (BN / 128), out);
}
